// WaveNet_86053964742872
// MI455X (gfx1250) — matrix operand order repaired and checked by reading, not compiled, not run on hardware
//
#include <hip/hip_runtime.h>

typedef __attribute__((ext_vector_type(16))) _Float16 v16h;
typedef __attribute__((ext_vector_type(8)))  _Float16 v8h;
typedef __attribute__((ext_vector_type(8)))  float    v8f;
typedef __attribute__((ext_vector_type(4)))  int      v4i;

constexpr int RCH   = 128;    // residual channels
constexpr int BATCH = 16;
constexpr int SLEN  = 4096;
constexpr int NT    = 32;     // positions per tile
constexpr int TPB   = SLEN / NT;        // tiles per batch row = 128
constexpr int GRID  = BATCH * TPB;      // 2048 workgroups
constexpr int LDSB  = 144;    // halves per LDS activation row (32B-aligned rows)
constexpr int LDSH  = 264;    // floats per LDS H row

// ---------------- async global->LDS staging (CDNA5) ----------------

#if defined(__has_builtin)
#if __has_builtin(__builtin_amdgcn_global_load_async_to_lds_b128) && \
    __has_builtin(__builtin_amdgcn_s_wait_asynccnt)
#define WN_ASYNC 1
#endif
#endif

// copy one 16-byte chunk global->LDS; async DMA when available
__device__ __forceinline__ void stage16(_Float16* lds, const _Float16* g) {
#ifdef WN_ASYNC
  __builtin_amdgcn_global_load_async_to_lds_b128(
      (__attribute__((address_space(1))) v4i*)(v4i*)const_cast<_Float16*>(g),
      (__attribute__((address_space(3))) v4i*)(v4i*)lds, 0, 0);
#else
  *(v8h*)lds = *(const v8h*)g;
#endif
}

__device__ __forceinline__ void stage_wait() {
#ifdef WN_ASYNC
  __builtin_amdgcn_s_wait_asynccnt(0);
#endif
}

// ---------------- WMMA helpers ----------------

__device__ __forceinline__ v8f wmma16(v16h a, v16h b, v8f c) {
  // D = A(16x32 f16) * B(32x16 f16) + C(16x16 f32)
  return __builtin_amdgcn_wmma_f32_16x16x32_f16(false, a, false, b, (short)0, c,
                                                false, false);
}

// A fragment: weights pre-packed into per-lane fragment order -> one 32B load.
__device__ __forceinline__ v16h load_a_frag(const _Float16* packed, int mt,
                                            int kt, int ktiles, int lane) {
  return *(const v16h*)(packed + ((size_t)(mt * ktiles + kt) * 32 + lane) * 16);
}

// B fragment from LDS tile stored [pos][channel] (row stride LDSB halves).
__device__ __forceinline__ v16h load_b_frag(const _Float16* lds, int n0, int k0,
                                            int lane) {
  int n = n0 + (lane & 15);
  int k = k0 + ((lane >> 4) << 3);
  v8h lo = *(const v8h*)(lds + n * LDSB + k);
  v8h hi = *(const v8h*)(lds + n * LDSB + k + 16);
  return __builtin_shufflevector(lo, hi, 0, 1, 2, 3, 4, 5, 6, 7, 8, 9, 10, 11, 12, 13, 14, 15);
}

// fragment (m,k) mapping for 16-bit A 16x32 tiles (ISA 7.12.2)
__device__ __forceinline__ int frag_k(int lane, int e) {
  return (e & 7) + ((e >> 3) << 4) + ((lane >> 4) << 3);
}

__device__ __forceinline__ float sigm(float x) { return 1.f / (1.f + expf(-x)); }

// ---------------- weight packing ----------------

// dw: [15][256][128][2] f32 -> per (layer,tap) packed 256x128 f16 fragments
__global__ void pack_wd(const float* __restrict__ dw, _Float16* __restrict__ dst) {
  int idx = blockIdx.x * blockDim.x + threadIdx.x;
  if (idx >= 15 * 65536) return;
  int e = idx & 15, lane = (idx >> 4) & 31, kt = (idx >> 9) & 3;
  int mt = (idx >> 11) & 15, t = (idx >> 15) & 1, i = idx >> 16;
  int m = mt * 16 + (lane & 15);
  int k = kt * 32 + frag_k(lane, e);
  dst[idx] = (_Float16)dw[(((size_t)i * 256 + m) * 128 + k) * 2 + t];
}

// src: [nmat][128][128] f32 -> packed 128x128 f16 fragments
__global__ void pack_m128(const float* __restrict__ src, _Float16* __restrict__ dst,
                          int nmat) {
  int idx = blockIdx.x * blockDim.x + threadIdx.x;
  if (idx >= nmat * 16384) return;
  int e = idx & 15, lane = (idx >> 4) & 31, kt = (idx >> 9) & 3;
  int mt = (idx >> 11) & 7, mat = idx >> 14;
  int m = mt * 16 + (lane & 15);
  int k = kt * 32 + frag_k(lane, e);
  dst[idx] = (_Float16)src[((size_t)mat * 128 + m) * 128 + k];
}

// fw_cond: [256][32][1] f32 -> packed 256x32 f16 fragments
__global__ void pack_wc(const float* __restrict__ src, _Float16* __restrict__ dst) {
  int idx = blockIdx.x * blockDim.x + threadIdx.x;
  if (idx >= 8192) return;
  int e = idx & 15, lane = (idx >> 4) & 31, mt = idx >> 9;
  int m = mt * 16 + (lane & 15);
  int k = frag_k(lane, e);
  dst[idx] = (_Float16)src[m * 32 + k];
}

// ---------------- first block: cond GEMM + 1-ch dilated conv + gate + skip ----
// activations are stored position-major: res[b][l][c] f16

__global__ void __launch_bounds__(256) wn_first(
    const float* __restrict__ x, const float* __restrict__ c,
    const _Float16* __restrict__ Wc, const float* __restrict__ fw_dil,
    const float* __restrict__ fw_dilb, const float* __restrict__ fw_condb,
    const _Float16* __restrict__ Wk, const float* __restrict__ fw_skipb,
    _Float16* __restrict__ res_out, _Float16* __restrict__ skips_out) {
  __shared__ __align__(32) _Float16 ct[NT * LDSB];
  __shared__ __align__(16) float Hs[NT * LDSH];
  __shared__ __align__(32) _Float16 Gs[NT * LDSB];

  const int tid = threadIdx.x;
  const int b = blockIdx.x / TPB;
  const int l0 = (blockIdx.x % TPB) * NT;
  const int wave = tid >> 5, lane = tid & 31;

  // stage condition tile [n][ch] f16 (32 channels, transpose from [B][32][L] f32)
  {
    int ch = tid >> 3, n0 = (tid & 7) * 4;
    const float* crow = c + ((size_t)b * 32 + ch) * SLEN + l0 + n0;
#pragma unroll
    for (int j = 0; j < 4; ++j) ct[(n0 + j) * LDSB + ch] = (_Float16)crow[j];
  }
  __syncthreads();

  // H = Wc[256x32] @ Ctile[32x32]
  v8f acc[2][2] = {};
  {
    v16h b0 = load_b_frag(ct, 0, 0, lane);
    v16h b1 = load_b_frag(ct, 16, 0, lane);
#pragma unroll
    for (int mi = 0; mi < 2; ++mi) {
      v16h a = load_a_frag(Wc, 2 * wave + mi, 0, 1, lane);
      acc[mi][0] = wmma16(a, b0, acc[mi][0]);
      acc[mi][1] = wmma16(a, b1, acc[mi][1]);
    }
  }
  // epilogue into LDS: + biases + 1-channel dilated conv on x (dil=1, K=2)
  const int mwave = wave * 32;
#pragma unroll
  for (int mi = 0; mi < 2; ++mi)
#pragma unroll
    for (int ni = 0; ni < 2; ++ni) {
      int n = ni * 16 + (lane & 15);
      int mb = mwave + mi * 16 + ((lane >> 4) << 3);
      int gl = l0 + n;
      float xcur = x[(size_t)b * SLEN + gl];
      float xprv = (gl > 0) ? x[(size_t)b * SLEN + gl - 1] : 0.f;
#pragma unroll
      for (int r = 0; r < 8; ++r) {
        int o = mb + r;
        Hs[n * LDSH + o] = acc[mi][ni][r] + fw_dilb[o] + fw_condb[o] +
                           fw_dil[o * 2] * xprv + fw_dil[o * 2 + 1] * xcur;
      }
    }
  __syncthreads();

  // gate: tanh(f) * sigmoid(g)
  {
    int n = tid >> 3, c0 = (tid & 7) * 16;
#pragma unroll
    for (int j = 0; j < 16; ++j) {
      float f = Hs[n * LDSH + c0 + j];
      float g = Hs[n * LDSH + 128 + c0 + j];
      Gs[n * LDSB + c0 + j] = (_Float16)(tanhf(f) * sigm(g));
    }
  }
  __syncthreads();

  // skip = Wk[128x128] @ G ; res = skip ; skips = skip  (position-major out)
  v8f s[2] = {};
#pragma unroll
  for (int kt = 0; kt < 4; ++kt) {
    v16h ba = load_b_frag(Gs, 0, kt * 32, lane);
    v16h bb = load_b_frag(Gs, 16, kt * 32, lane);
    v16h a = load_a_frag(Wk, wave, kt, 4, lane);
    s[0] = wmma16(a, ba, s[0]);
    s[1] = wmma16(a, bb, s[1]);
  }
#pragma unroll
  for (int ni = 0; ni < 2; ++ni) {
    int n = ni * 16 + (lane & 15);
    int mb = wave * 16 + ((lane >> 4) << 3);
    size_t gi = ((size_t)b * SLEN + l0 + n) * RCH + mb;
    v8h ro;
#pragma unroll
    for (int r = 0; r < 8; ++r) ro[r] = (_Float16)(s[ni][r] + fw_skipb[mb + r]);
    *(v8h*)(res_out + gi) = ro;
    *(v8h*)(skips_out + gi) = ro;
  }
}

// ---------------- dilated residual block ----------------

__global__ void __launch_bounds__(256) wn_block(
    const _Float16* __restrict__ res_in, _Float16* __restrict__ res_out,
    _Float16* __restrict__ skips, const _Float16* __restrict__ Wd0,
    const _Float16* __restrict__ Wd1, const float* __restrict__ db,
    const _Float16* __restrict__ Ws, const float* __restrict__ sb, int dil) {
  __shared__ __align__(32) _Float16 bt0[NT * LDSB];  // res(t)
  __shared__ __align__(32) _Float16 bt1[NT * LDSB];  // res(t - dil)
  __shared__ __align__(16) float Hs[NT * LDSH];
  __shared__ __align__(32) _Float16 Gs[NT * LDSB];

  const int tid = threadIdx.x;
  const int b = blockIdx.x / TPB;
  const int l0 = (blockIdx.x % TPB) * NT;
  const int wave = tid >> 5, lane = tid & 31;
  const int mwave = wave * 32;

  // pull the weight streams toward us (global_prefetch_b8)
  if (lane == 0) {
    __builtin_prefetch(Wd0 + (size_t)mwave * 128, 0, 1);
    __builtin_prefetch(Wd1 + (size_t)mwave * 128, 0, 1);
    __builtin_prefetch(Ws + (size_t)wave * 16 * 128, 0, 1);
  }

  // stage activation tiles [n][ch]: contiguous rows in position-major global,
  // DMA'd straight into LDS (ASYNCcnt); causal left edge is zero-filled.
  {
    int n = tid >> 3;  // 32 rows, 8 threads per row
    const _Float16* row0 = res_in + ((size_t)b * SLEN + l0 + n) * RCH;
    int gp = l0 + n - dil;
#pragma unroll
    for (int q = 0; q < 2; ++q) {
      int ch = ((tid & 7) * 2 + q) * 8;  // 16B chunk of channels
      stage16(&bt0[n * LDSB + ch], row0 + ch);
      if (gp >= 0) {
        stage16(&bt1[n * LDSB + ch],
                res_in + ((size_t)b * SLEN + gp) * RCH + ch);
      } else {
        v8h z = {};
        *(v8h*)&bt1[n * LDSB + ch] = z;
      }
    }
  }
  stage_wait();
  __syncthreads();

  // H[256x32] = W1 @ res(t) + W0 @ res(t-d)   (K = 128)
  v8f acc[2][2] = {};
#pragma unroll
  for (int kt = 0; kt < 4; ++kt) {
    v16h b0a = load_b_frag(bt0, 0, kt * 32, lane);
    v16h b0b = load_b_frag(bt0, 16, kt * 32, lane);
    v16h b1a = load_b_frag(bt1, 0, kt * 32, lane);
    v16h b1b = load_b_frag(bt1, 16, kt * 32, lane);
#pragma unroll
    for (int mi = 0; mi < 2; ++mi) {
      v16h a1 = load_a_frag(Wd1, 2 * wave + mi, kt, 4, lane);
      v16h a0 = load_a_frag(Wd0, 2 * wave + mi, kt, 4, lane);
      acc[mi][0] = wmma16(a1, b0a, acc[mi][0]);
      acc[mi][1] = wmma16(a1, b0b, acc[mi][1]);
      acc[mi][0] = wmma16(a0, b1a, acc[mi][0]);
      acc[mi][1] = wmma16(a0, b1b, acc[mi][1]);
    }
  }
#pragma unroll
  for (int mi = 0; mi < 2; ++mi)
#pragma unroll
    for (int ni = 0; ni < 2; ++ni) {
      int n = ni * 16 + (lane & 15);
      int mb = mwave + mi * 16 + ((lane >> 4) << 3);
#pragma unroll
      for (int r = 0; r < 8; ++r)
        Hs[n * LDSH + mb + r] = acc[mi][ni][r] + db[mb + r];
    }
  __syncthreads();

  // gate
  {
    int n = tid >> 3, c0 = (tid & 7) * 16;
#pragma unroll
    for (int j = 0; j < 16; ++j) {
      float f = Hs[n * LDSH + c0 + j];
      float g = Hs[n * LDSH + 128 + c0 + j];
      Gs[n * LDSB + c0 + j] = (_Float16)(tanhf(f) * sigm(g));
    }
  }
  __syncthreads();

  // skip = Ws[128x128] @ G ; res_out = res_in + skip ; skips += skip
  v8f s[2] = {};
#pragma unroll
  for (int kt = 0; kt < 4; ++kt) {
    v16h ba = load_b_frag(Gs, 0, kt * 32, lane);
    v16h bb = load_b_frag(Gs, 16, kt * 32, lane);
    v16h a = load_a_frag(Ws, wave, kt, 4, lane);
    s[0] = wmma16(a, ba, s[0]);
    s[1] = wmma16(a, bb, s[1]);
  }
#pragma unroll
  for (int ni = 0; ni < 2; ++ni) {
    int n = ni * 16 + (lane & 15);
    int mb = wave * 16 + ((lane >> 4) << 3);
    size_t gi = ((size_t)b * SLEN + l0 + n) * RCH + mb;
    v8h rv = *(const v8h*)(res_in + gi);
    v8h kv = *(const v8h*)(skips + gi);
    v8h ro, ko;
#pragma unroll
    for (int r = 0; r < 8; ++r) {
      float sv = s[ni][r] + sb[mb + r];
      ro[r] = (_Float16)((float)rv[r] + sv);
      ko[r] = (_Float16)((float)kv[r] + sv);
    }
    *(v8h*)(res_out + gi) = ro;
    *(v8h*)(skips + gi) = ko;
  }
}

// ---------------- output head: relu -> 1x1(128) -> relu -> 1x1(1) -----------

__global__ void __launch_bounds__(256) wn_out(
    const _Float16* __restrict__ skips, const _Float16* __restrict__ W1,
    const float* __restrict__ ob1, const float* __restrict__ ow2,
    const float* __restrict__ ob2, float* __restrict__ out) {
  __shared__ __align__(32) _Float16 st[NT * LDSB];
  __shared__ __align__(16) float Hs[NT * LDSH];
  __shared__ float red[8][32];

  const int tid = threadIdx.x;
  const int b = blockIdx.x / TPB;
  const int l0 = (blockIdx.x % TPB) * NT;
  const int wave = tid >> 5, lane = tid & 31;

  // DMA skips tile into LDS, then apply ReLU in place
  {
    int n = tid >> 3;
    const _Float16* row = skips + ((size_t)b * SLEN + l0 + n) * RCH;
#pragma unroll
    for (int q = 0; q < 2; ++q) {
      int ch = ((tid & 7) * 2 + q) * 8;
      stage16(&st[n * LDSB + ch], row + ch);
    }
  }
  stage_wait();
  __syncthreads();
  {
    int n = tid >> 3, c0 = (tid & 7) * 16;
#pragma unroll
    for (int j = 0; j < 16; ++j) {
      float f = (float)st[n * LDSB + c0 + j];
      st[n * LDSB + c0 + j] = (_Float16)(f > 0.f ? f : 0.f);
    }
  }
  __syncthreads();

  // h = relu(W1[128x128] @ tile + ob1)
  v8f s[2] = {};
#pragma unroll
  for (int kt = 0; kt < 4; ++kt) {
    v16h ba = load_b_frag(st, 0, kt * 32, lane);
    v16h bb = load_b_frag(st, 16, kt * 32, lane);
    v16h a = load_a_frag(W1, wave, kt, 4, lane);
    s[0] = wmma16(a, ba, s[0]);
    s[1] = wmma16(a, bb, s[1]);
  }
#pragma unroll
  for (int ni = 0; ni < 2; ++ni) {
    int n = ni * 16 + (lane & 15);
    int mb = wave * 16 + ((lane >> 4) << 3);
#pragma unroll
    for (int r = 0; r < 8; ++r) {
      float v = s[ni][r] + ob1[mb + r];
      Hs[n * LDSH + mb + r] = v > 0.f ? v : 0.f;
    }
  }
  __syncthreads();

  // out[t] = ob2 + ow2 . h[:,t]   (8-way partial reduce)
  {
    int n = tid & 31, part = tid >> 5;
    float p = 0.f;
#pragma unroll
    for (int j = 0; j < 16; ++j)
      p += ow2[part * 16 + j] * Hs[n * LDSH + part * 16 + j];
    red[part][n] = p;
  }
  __syncthreads();
  if (tid < 32) {
    float sum = ob2[0];
#pragma unroll
    for (int q = 0; q < 8; ++q) sum += red[q][tid];
    out[(size_t)b * SLEN + l0 + tid] = sum;
  }
}

// ---------------- host orchestration ----------------

extern "C" void kernel_launch(void* const* d_in, const int* in_sizes, int n_in,
                              void* d_out, int out_size, void* d_ws, size_t ws_size,
                              hipStream_t stream) {
  const float* x        = (const float*)d_in[0];
  const float* c        = (const float*)d_in[1];
  const float* fw_dil   = (const float*)d_in[2];
  const float* fw_dilb  = (const float*)d_in[3];
  const float* fw_cond  = (const float*)d_in[4];
  const float* fw_condb = (const float*)d_in[5];
  const float* fw_skip  = (const float*)d_in[6];
  const float* fw_skipb = (const float*)d_in[7];
  const float* dw       = (const float*)d_in[8];
  const float* db       = (const float*)d_in[9];
  const float* sw       = (const float*)d_in[10];
  const float* sb       = (const float*)d_in[11];
  const float* ow1      = (const float*)d_in[12];
  const float* ob1      = (const float*)d_in[13];
  const float* ow2      = (const float*)d_in[14];
  const float* ob2      = (const float*)d_in[15];
  float* out = (float*)d_out;

  const size_t act = (size_t)BATCH * RCH * SLEN;  // 8,388,608 halves (16MB)
  _Float16* resA = (_Float16*)d_ws;
  _Float16* resB = resA + act;
  _Float16* skp  = resB + act;
  _Float16* WD   = skp + act;                 // 15*2*256*128
  _Float16* WS_  = WD + (size_t)15 * 2 * 256 * 128;
  _Float16* WC   = WS_ + (size_t)15 * 128 * 128;
  _Float16* WK   = WC + 256 * 32;
  _Float16* WO1  = WK + 128 * 128;

  // pack all weights to f16 WMMA fragment order
  pack_wd <<<(15 * 65536 + 255) / 256, 256, 0, stream>>>(dw, WD);
  pack_m128<<<(15 * 16384 + 255) / 256, 256, 0, stream>>>(sw, WS_, 15);
  pack_m128<<<(16384 + 255) / 256, 256, 0, stream>>>(fw_skip, WK, 1);
  pack_m128<<<(16384 + 255) / 256, 256, 0, stream>>>(ow1, WO1, 1);
  pack_wc  <<<(8192 + 255) / 256, 256, 0, stream>>>(fw_cond, WC);

  // first block
  wn_first<<<GRID, 256, 0, stream>>>(x, c, WC, fw_dil, fw_dilb, fw_condb, WK,
                                     fw_skipb, resA, skp);

  // 15 dilated residual blocks (DILS[1:])
  static const int dils[15] = {2, 4, 8, 16, 32, 64, 128, 1, 2, 4, 8, 16, 32, 64, 128};
  _Float16* cur = resA;
  _Float16* nxt = resB;
  for (int i = 0; i < 15; ++i) {
    wn_block<<<GRID, 256, 0, stream>>>(
        cur, nxt, skp,
        WD + (size_t)(2 * i) * 32768,      // tap0 (t - d)
        WD + (size_t)(2 * i + 1) * 32768,  // tap1 (t)
        db + (size_t)i * 256,
        WS_ + (size_t)i * 16384,
        sb + (size_t)i * 128,
        dils[i]);
    _Float16* t = cur; cur = nxt; nxt = t;
  }

  // output head
  wn_out<<<GRID, 256, 0, stream>>>(skp, WO1, ob1, ow2, ob2, out);
}
